// Seq2Seq_27023934227014
// MI455X (gfx1250) — compile-verified, matrix operand order checked by reading, not run on hardware
//
#include <hip/hip_runtime.h>

// ---------------------------------------------------------------------------
// Types for CDNA5 WMMA (wave32, V_WMMA_F32_16X16X32_BF16)
// ---------------------------------------------------------------------------
typedef __bf16 bf16;
typedef __bf16 v16bf __attribute__((ext_vector_type(16)));
typedef __bf16 v8bf  __attribute__((ext_vector_type(8)));
typedef float  v8f   __attribute__((ext_vector_type(8)));

#define WMMA_BF16(a, b, c) \
  __builtin_amdgcn_wmma_f32_16x16x32_bf16(false, (a), false, (b), (short)0, (c), false, false)

// Load one 16x32 bf16 operand fragment (A row-major 16 rows x K cols, or
// B as W^T where W is row-major N x K: identical access pattern).
// Per ISA 7.12.2: lane<16 -> row = lane, elems 0..7 = K[k..k+7], 8..15 = K[k+16..k+23]
//                 lane>=16 -> same row, elems 0..7 = K[k+8..k+15], 8..15 = K[k+24..k+31]
__device__ __forceinline__ v16bf load_frag(const bf16* base, int ld, int kbase, int lane) {
  int r    = lane & 15;
  int koff = (lane >> 4) << 3;
  const bf16* p = base + (size_t)r * ld + kbase + koff;
  v8bf lo = *(const v8bf*)(p);
  v8bf hi = *(const v8bf*)(p + 16);
  v16bf v;
#pragma unroll
  for (int i = 0; i < 8; i++) { v[i] = lo[i]; v[8 + i] = hi[i]; }
  return v;
}

// ---------------------------------------------------------------------------
// fp32 -> bf16 conversion
// ---------------------------------------------------------------------------
__global__ void cvt_bf16_kernel(const float* __restrict__ in, bf16* __restrict__ out, int n) {
  for (int i = blockIdx.x * blockDim.x + threadIdx.x; i < n; i += gridDim.x * blockDim.x)
    out[i] = (bf16)in[i];
}

// ---------------------------------------------------------------------------
// Embedding gather + bf16 convert.  X[row = s*Bsz + b, :] = emb[tok[b*Steps+s], :]
// ---------------------------------------------------------------------------
__global__ void gather_embed_kernel(const float* __restrict__ emb, const int* __restrict__ tok,
                                    int Bsz, int Steps, bf16* __restrict__ X, int E) {
  int row = blockIdx.x;
  int s = row / Bsz, b = row % Bsz;
  int t = tok[b * Steps + s];
  const float* srcp = emb + (size_t)t * E;
  bf16* dst = X + (size_t)row * E;
  for (int i = threadIdx.x; i < E; i += blockDim.x) dst[i] = (bf16)srcp[i];
}

// ---------------------------------------------------------------------------
// Generic bf16 WMMA GEMM:  C[m,n] = sum_k A[m,k] * W[n,k]  (+bias, optional relu)
// A: M x K bf16 row-major.  W: N x K bf16 row-major.  Each wave: 16x64 C-tile.
// Cf: optional f32 out (optionally permuted (t*B+b,n) -> out[b,t,n]).
// Cb: optional bf16 out (row-major).
// ---------------------------------------------------------------------------
__global__ __launch_bounds__(256) void gemm_wmma_kernel(
    const bf16* __restrict__ A, const bf16* __restrict__ W,
    int M, int N, int K,
    const float* __restrict__ bias, int relu,
    float* __restrict__ Cf, bf16* __restrict__ Cb,
    int permBT, int Bsz, int Tlen)
{
  int lane = threadIdx.x & 31;
  int wave = blockIdx.x * (blockDim.x >> 5) + (threadIdx.x >> 5);
  int mTiles = M >> 4;
  int nGroups = N >> 6;
  if (wave >= mTiles * nGroups) return;
  int mt = wave % mTiles;
  int ng = wave / mTiles;
  int m0 = mt << 4;
  int n0 = ng << 6;

  v8f acc0, acc1, acc2, acc3;
#pragma unroll
  for (int r = 0; r < 8; r++) { acc0[r] = 0.f; acc1[r] = 0.f; acc2[r] = 0.f; acc3[r] = 0.f; }

  const bf16* Ab = A + (size_t)m0 * K;
  const bf16* W0 = W + (size_t)(n0 +  0) * K;
  const bf16* W1 = W + (size_t)(n0 + 16) * K;
  const bf16* W2 = W + (size_t)(n0 + 32) * K;
  const bf16* W3 = W + (size_t)(n0 + 48) * K;

#pragma unroll 4
  for (int k = 0; k < K; k += 32) {
    v16bf a  = load_frag(Ab, K, k, lane);
    v16bf b0 = load_frag(W0, K, k, lane);
    v16bf b1 = load_frag(W1, K, k, lane);
    v16bf b2 = load_frag(W2, K, k, lane);
    v16bf b3 = load_frag(W3, K, k, lane);
    acc0 = WMMA_BF16(a, b0, acc0);
    acc1 = WMMA_BF16(a, b1, acc1);
    acc2 = WMMA_BF16(a, b2, acc2);
    acc3 = WMMA_BF16(a, b3, acc3);
  }

  int colL   = lane & 15;
  int rowOff = (lane >> 4) << 3;
  v8f accs[4] = {acc0, acc1, acc2, acc3};
#pragma unroll
  for (int t = 0; t < 4; t++) {
    int col = n0 + t * 16 + colL;
    float bv = bias ? bias[col] : 0.f;
#pragma unroll
    for (int r = 0; r < 8; r++) {
      int row = m0 + rowOff + r;
      float v = accs[t][r] + bv;
      if (relu) v = v > 0.f ? v : 0.f;
      if (Cf) {
        size_t idx;
        if (permBT) { int bb = row % Bsz; int tt = row / Bsz;
                      idx = ((size_t)bb * Tlen + tt) * (size_t)N + col; }
        else idx = (size_t)row * N + col;
        Cf[idx] = v;
      }
      if (Cb) Cb[(size_t)row * N + col] = (bf16)v;
    }
  }
}

// ---------------------------------------------------------------------------
// Persistent LSTM recurrence. One 1024-thread workgroup (32 wave32) per
// direction (gridDim.x = 2).  Per step:
//   gates(32x2048) = GX[step] + h @ Whh^T + (bih+bhh)   -- 256 WMMA tiles
//   nonlinearity + state update; h kept bf16 in LDS, c f32 in registers.
// LDS: G gates 32x2048 bf16 (128KB) + h 32x512 bf16 (32KB) = 160KB of 320KB.
// ---------------------------------------------------------------------------
struct LstmDir {
  const bf16* GX;          // (steps*32, 2048) bf16, precomputed x@Wih^T
  const bf16* Whh;         // (2048, 512) bf16
  const float* bih; const float* bhh;
  const float* h0; const float* c0;   // (32,512) f32 or null -> zeros
  int reverse;             // 1: step s consumes GX rows (steps-1-s)*32..
  float* hOut; float* cOut;           // final states, nullable
  bf16* hSeq; int hSeqLd; int hSeqOff; // per-step h store, nullable
};

__global__ __launch_bounds__(1024) void lstm_seq_kernel(LstmDir d0, LstmDir d1, int steps) {
  extern __shared__ char smem[];
  bf16* G  = (bf16*)smem;                       // 32 x 2048
  bf16* hS = (bf16*)(smem + 32 * 2048 * 2);     // 32 x 512

  LstmDir A = (blockIdx.x == 0) ? d0 : d1;
  int tid  = threadIdx.x;
  int lane = tid & 31;
  int wave = tid >> 5;

  // per-thread element slice of the 32x512 state: b = tid>>5, j = (tid&31)*16 + u
  int sb = tid >> 5;
  int sj = (tid & 31) << 4;
  float cReg[16];
#pragma unroll
  for (int u = 0; u < 16; u++) {
    int idx = sb * 512 + sj + u;
    cReg[u] = A.c0 ? A.c0[idx] : 0.f;
    hS[idx] = (bf16)(A.h0 ? A.h0[idx] : 0.f);
  }
  __syncthreads();

  int mt  = wave & 1;      // M-tile (rows 0-15 / 16-31)
  int grp = wave >> 1;     // 16 groups x 8 N-tiles = 128 N-tiles
  int m0  = mt << 4;
  int colL   = lane & 15;
  int rowOff = (lane >> 4) << 3;

  for (int s = 0; s < steps; s++) {
    int gr = (A.reverse ? (steps - 1 - s) : s) * 32;
    const bf16* GXrow = A.GX + (size_t)(gr + m0) * 2048;
    const bf16* Ab = hS + m0 * 512;

#pragma unroll 1
    for (int j = 0; j < 8; j++) {
      int n0  = (grp * 8 + j) << 4;
      int col = n0 + colL;
      float bsum = A.bih[col] + A.bhh[col];
      v8f acc;
#pragma unroll
      for (int r = 0; r < 8; r++)
        acc[r] = (float)GXrow[(size_t)(rowOff + r) * 2048 + col] + bsum;

      const bf16* Wb = A.Whh + (size_t)n0 * 512;
#pragma unroll 4
      for (int k = 0; k < 512; k += 32) {
        v16bf af = load_frag(Ab, 512, k, lane);
        v16bf bf = load_frag(Wb, 512, k, lane);
        acc = WMMA_BF16(af, bf, acc);
      }
#pragma unroll
      for (int r = 0; r < 8; r++)
        G[(m0 + rowOff + r) * 2048 + col] = (bf16)acc[r];
    }
    __syncthreads();

    // elementwise gate math (gate order i,f,g,o across the 2048 dim)
#pragma unroll
    for (int u = 0; u < 16; u++) {
      int j = sj + u;
      float ig = (float)G[sb * 2048 +        j];
      float fg = (float)G[sb * 2048 +  512 + j];
      float gg = (float)G[sb * 2048 + 1024 + j];
      float og = (float)G[sb * 2048 + 1536 + j];
      ig = 1.f / (1.f + __expf(-ig));
      fg = 1.f / (1.f + __expf(-fg));
      gg = tanhf(gg);
      og = 1.f / (1.f + __expf(-og));
      float c = fg * cReg[u] + ig * gg;
      cReg[u] = c;
      float h = og * tanhf(c);
      hS[sb * 512 + j] = (bf16)h;
      if (A.hSeq)
        A.hSeq[(size_t)(s * 32 + sb) * A.hSeqLd + A.hSeqOff + j] = (bf16)h;
    }
    __syncthreads();
  }

  if (A.hOut) {
#pragma unroll
    for (int u = 0; u < 16; u++) A.hOut[sb * 512 + sj + u] = (float)hS[sb * 512 + sj + u];
  }
  if (A.cOut) {
#pragma unroll
    for (int u = 0; u < 16; u++) A.cOut[sb * 512 + sj + u] = cReg[u];
  }
}

// ---------------------------------------------------------------------------
// Host orchestration
// ---------------------------------------------------------------------------
extern "C" void kernel_launch(void* const* d_in, const int* in_sizes, int n_in,
                              void* d_out, int out_size, void* d_ws, size_t ws_size,
                              hipStream_t stream) {
  (void)in_sizes; (void)n_in; (void)out_size; (void)ws_size;
  const int Bsz = 32, S = 128, T = 64, E = 512, H = 512, H4 = 2048, V = 32000;

  const int*   src      = (const int*)  d_in[0];
  const int*   trg      = (const int*)  d_in[1];
  // d_in[2] = teacher_ratio == 1 -> teacher forcing always on (uniform[0,1) < 1)
  const float* enc_emb  = (const float*)d_in[3];
  const float* eWihF    = (const float*)d_in[4];
  const float* eWhhF    = (const float*)d_in[5];
  const float* eBihF    = (const float*)d_in[6];
  const float* eBhhF    = (const float*)d_in[7];
  const float* eWihB    = (const float*)d_in[8];
  const float* eWhhB    = (const float*)d_in[9];
  const float* eBihB    = (const float*)d_in[10];
  const float* eBhhB    = (const float*)d_in[11];
  const float* dec_emb  = (const float*)d_in[12];
  const float* dWihF    = (const float*)d_in[13];
  const float* dWhhF    = (const float*)d_in[14];
  const float* dBihF    = (const float*)d_in[15];
  const float* dBhhF    = (const float*)d_in[16];
  const float* dWihB    = (const float*)d_in[17];
  const float* dWhhB    = (const float*)d_in[18];
  const float* dBihB    = (const float*)d_in[19];
  const float* dBhhB    = (const float*)d_in[20];
  const float* w1       = (const float*)d_in[21];
  const float* b1       = (const float*)d_in[22];
  const float* w2       = (const float*)d_in[23];
  const float* b2       = (const float*)d_in[24];

  // workspace bump allocator (256B aligned)
  char* ws = (char*)d_ws;
  size_t off = 0;
  auto alloc = [&](size_t bytes) -> char* {
    char* p = ws + off;
    off += (bytes + 255) & ~(size_t)255;
    return p;
  };

  bf16* eWihF_b = (bf16*)alloc((size_t)H4 * E * 2);
  bf16* eWhhF_b = (bf16*)alloc((size_t)H4 * H * 2);
  bf16* eWihB_b = (bf16*)alloc((size_t)H4 * E * 2);
  bf16* eWhhB_b = (bf16*)alloc((size_t)H4 * H * 2);
  bf16* dWihF_b = (bf16*)alloc((size_t)H4 * E * 2);
  bf16* dWhhF_b = (bf16*)alloc((size_t)H4 * H * 2);
  bf16* dWihB_b = (bf16*)alloc((size_t)H4 * E * 2);
  bf16* dWhhB_b = (bf16*)alloc((size_t)H4 * H * 2);
  bf16* w1_b    = (bf16*)alloc((size_t)H * 2 * H * 2);
  bf16* w2_b    = (bf16*)alloc((size_t)V * H * 2);
  bf16* Xenc    = (bf16*)alloc((size_t)S * Bsz * E * 2);
  bf16* Xdec    = (bf16*)alloc((size_t)T * Bsz * E * 2);
  bf16* GXeF    = (bf16*)alloc((size_t)S * Bsz * H4 * 2);
  bf16* GXeB    = (bf16*)alloc((size_t)S * Bsz * H4 * 2);
  bf16* GXdF    = (bf16*)alloc((size_t)T * Bsz * H4 * 2);
  bf16* GXdB    = (bf16*)alloc((size_t)T * Bsz * H4 * 2);
  float* hf     = (float*)alloc((size_t)Bsz * H * 4);
  float* cf     = (float*)alloc((size_t)Bsz * H * 4);
  float* hb     = (float*)alloc((size_t)Bsz * H * 4);
  float* cb     = (float*)alloc((size_t)Bsz * H * 4);
  bf16* Zin     = (bf16*)alloc((size_t)T * Bsz * 2 * H * 2);
  bf16* zbuf    = (bf16*)alloc((size_t)T * Bsz * H * 2);

  auto cvt = [&](const float* in, bf16* outp, int n) {
    cvt_bf16_kernel<<<(n + 255) / 256, 256, 0, stream>>>(in, outp, n);
  };
  cvt(eWihF, eWihF_b, H4 * E);  cvt(eWhhF, eWhhF_b, H4 * H);
  cvt(eWihB, eWihB_b, H4 * E);  cvt(eWhhB, eWhhB_b, H4 * H);
  cvt(dWihF, dWihF_b, H4 * E);  cvt(dWhhF, dWhhF_b, H4 * H);
  cvt(dWihB, dWihB_b, H4 * E);  cvt(dWhhB, dWhhB_b, H4 * H);
  cvt(w1, w1_b, H * 2 * H);     cvt(w2, w2_b, V * H);

  gather_embed_kernel<<<S * Bsz, 256, 0, stream>>>(enc_emb, src, Bsz, S, Xenc, E);
  gather_embed_kernel<<<T * Bsz, 256, 0, stream>>>(dec_emb, trg, Bsz, T, Xdec, E);

  auto gemm = [&](const bf16* A, const bf16* W, int M, int N, int K,
                  const float* bias, int relu, float* Cf_, bf16* Cb_, int permBT) {
    int jobs = (M >> 4) * (N >> 6);
    gemm_wmma_kernel<<<(jobs + 7) / 8, 256, 0, stream>>>(
        A, W, M, N, K, bias, relu, Cf_, Cb_, permBT, Bsz, T);
  };

  // batched input projections x @ Wih^T for all timesteps
  gemm(Xenc, eWihF_b, S * Bsz, H4, E, nullptr, 0, nullptr, GXeF, 0);
  gemm(Xenc, eWihB_b, S * Bsz, H4, E, nullptr, 0, nullptr, GXeB, 0);
  gemm(Xdec, dWihF_b, T * Bsz, H4, E, nullptr, 0, nullptr, GXdF, 0);
  gemm(Xdec, dWihB_b, T * Bsz, H4, E, nullptr, 0, nullptr, GXdB, 0);

  // encoder recurrence (fwd + time-reversed bwd), final states only
  size_t smemBytes = (size_t)32 * 2048 * 2 + (size_t)32 * 512 * 2; // 160KB
  LstmDir e0 = {GXeF, eWhhF_b, eBihF, eBhhF, nullptr, nullptr, 0, hf, cf, nullptr, 0, 0};
  LstmDir e1 = {GXeB, eWhhB_b, eBihB, eBhhB, nullptr, nullptr, 1, hb, cb, nullptr, 0, 0};
  lstm_seq_kernel<<<2, 1024, smemBytes, stream>>>(e0, e1, S);

  // decoder recurrence: two cells, both forward in time, init from encoder finals;
  // per-step h written (bf16) directly into Zin = concat(hf, hb)
  LstmDir dd0 = {GXdF, dWhhF_b, dBihF, dBhhF, hf, cf, 0, nullptr, nullptr, Zin, 2 * H, 0};
  LstmDir dd1 = {GXdB, dWhhB_b, dBihB, dBhhB, hb, cb, 0, nullptr, nullptr, Zin, 2 * H, H};
  lstm_seq_kernel<<<2, 1024, smemBytes, stream>>>(dd0, dd1, T);

  // MLP head: z = relu(Zin @ w1^T + b1) (bf16), logits = z @ w2^T + b2 (f32, permuted to (B,T,V))
  gemm(Zin, w1_b, T * Bsz, H, 2 * H, b1, 1, nullptr, zbuf, 0);
  gemm(zbuf, w2_b, T * Bsz, V, H, b2, 0, (float*)d_out, nullptr, 1);
}
